// GPT_60748017435067
// MI455X (gfx1250) — matrix operand order repaired and checked by reading, not compiled, not run on hardware
//
#include <hip/hip_runtime.h>
#include <hip/hip_bf16.h>

// ---------------------------------------------------------------------------
// GPT-2 small forward on gfx1250 (CDNA5): bf16 WMMA everywhere.
//   L=12, C=768, H=12, D=64, B=4, T=1024, V=50257
// ---------------------------------------------------------------------------

typedef __bf16 bf16_t;
typedef __attribute__((ext_vector_type(16))) __bf16 v16bf;
typedef __attribute__((ext_vector_type(8)))  __bf16 v8bf;
typedef __attribute__((ext_vector_type(8)))  float  v8f;

#define Bc 4
#define Tc 1024
#define Cc 768
#define Hc 12
#define Dc 64
#define Lc 12
#define Vc 50257
#define BTc (Bc * Tc)

// ---------------------------------------------------------------------------
// WMMA helpers
// ---------------------------------------------------------------------------

__device__ __forceinline__ v8f wmma_bf16(v16bf a, v16bf b, v8f c) {
    // D = A(16x32 bf16) * B(32x16 bf16) + C(16x16 f32)
    return __builtin_amdgcn_wmma_f32_16x16x32_bf16(
        /*neg_a=*/false, a, /*neg_b=*/false, b,
        /*c_mod=*/(short)0, c, /*reuse_a=*/false, /*reuse_b=*/false);
}

__device__ __forceinline__ v16bf combine16(v8bf lo, v8bf hi) {
    return __builtin_shufflevector(lo, hi, 0, 1, 2, 3, 4, 5, 6, 7,
                                           8, 9, 10, 11, 12, 13, 14, 15);
}

// A fragment (16x32, M x K), row-major source with leading dim ld.
// Lanes 0-15 hold row m = lane, K = [0..7]++[16..23];
// lanes 16-31 hold K = [8..15]++[24..31].
__device__ __forceinline__ v16bf load_a_frag(const bf16_t* base, int ld, int lane) {
    int m  = lane & 15;
    int hi = lane >> 4;
    const bf16_t* p = base + (long)m * ld + hi * 8;
    v8bf lo  = *(const v8bf*)(p);
    v8bf hi8 = *(const v8bf*)(p + 16);
    return combine16(lo, hi8);
}

// B fragment (32x16, K x N) where source is W[N,K] row-major (so B = W^T).
// Lane j holds column n = j&15: K = [0..7]++[16..23] (lanes 0-15) or [8..15]++[24..31] (lanes 16-31).
__device__ __forceinline__ v16bf load_b_frag(const bf16_t* base, int ld, int lane) {
    int n  = lane & 15;
    int hi = lane >> 4;
    const bf16_t* p = base + (long)n * ld + hi * 8;
    v8bf lo  = *(const v8bf*)(p);
    v8bf hi8 = *(const v8bf*)(p + 16);
    return combine16(lo, hi8);
}

__device__ __forceinline__ v16bf zero_frag() {
    v16bf z;
#pragma unroll
    for (int i = 0; i < 16; ++i) z[i] = (bf16_t)0.0f;
    return z;
}

__device__ __forceinline__ v16bf load_b_frag_guard(const bf16_t* W, int n0, int Ntot,
                                                   int ld, int kb, int lane) {
    int n = n0 + (lane & 15);
    if (n >= Ntot) return zero_frag();
    return load_b_frag(W + (long)n0 * ld + kb, ld, lane);
}

// ---------------------------------------------------------------------------
// Generic GEMM: out[M,N] = A[M,K] (bf16) * W[N,K]^T (bf16), f32 accumulate.
// mode 0: store f32; mode 1: store bf16; mode 2: f32 residual add.
// One wave computes a 32(M) x 64(N) tile: 2 A-frags, 4 B-frags, 8 WMMAs / K-step.
// Full-width tiles take an unguarded fast path (only the last N-tile of the
// lm_head GEMM is ever partial).
// ---------------------------------------------------------------------------
__global__ void __launch_bounds__(32)
gemm_bf16_kernel(const bf16_t* __restrict__ A, const bf16_t* __restrict__ W,
                 void* __restrict__ out, int M, int N, int K, int mode) {
    const int m0   = blockIdx.x * 32;
    const int n0   = blockIdx.y * 64;
    const int lane = threadIdx.x;

    v8f acc[2][4];
#pragma unroll
    for (int mi = 0; mi < 2; ++mi)
#pragma unroll
        for (int c = 0; c < 4; ++c)
#pragma unroll
            for (int i = 0; i < 8; ++i) acc[mi][c][i] = 0.0f;

    const bool full = (n0 + 64 <= N);   // wave-uniform
    if (full) {
        const bf16_t* Arow0 = A + (long)m0 * K;
        const bf16_t* Arow1 = A + (long)(m0 + 16) * K;
        for (int kb = 0; kb < K; kb += 32) {
            v16bf a0 = load_a_frag(Arow0 + kb, K, lane);
            v16bf a1 = load_a_frag(Arow1 + kb, K, lane);
#pragma unroll
            for (int c = 0; c < 4; ++c) {
                v16bf bf = load_b_frag(W + (long)(n0 + c * 16) * K + kb, K, lane);
                acc[0][c] = wmma_bf16(a0, bf, acc[0][c]);
                acc[1][c] = wmma_bf16(a1, bf, acc[1][c]);
            }
        }
    } else {
        for (int kb = 0; kb < K; kb += 32) {
            v16bf a0 = load_a_frag(A + (long)m0 * K + kb, K, lane);
            v16bf a1 = load_a_frag(A + (long)(m0 + 16) * K + kb, K, lane);
#pragma unroll
            for (int c = 0; c < 4; ++c) {
                v16bf bf = load_b_frag_guard(W, n0 + c * 16, N, K, kb, lane);
                acc[0][c] = wmma_bf16(a0, bf, acc[0][c]);
                acc[1][c] = wmma_bf16(a1, bf, acc[1][c]);
            }
        }
    }

    // C/D layout: VGPR i -> row +i (lanes 0-15) / +i+8 (lanes 16-31), col = lane&15.
    const int rowoff = (lane >> 4) * 8;
    const int ncol   = lane & 15;
#pragma unroll
    for (int mi = 0; mi < 2; ++mi) {
#pragma unroll
        for (int c = 0; c < 4; ++c) {
#pragma unroll
            for (int i = 0; i < 8; ++i) {
                int r   = m0 + mi * 16 + i + rowoff;
                int col = n0 + c * 16 + ncol;
                if (col < N) {
                    long idx = (long)r * N + col;
                    float v  = acc[mi][c][i];
                    if (mode == 0)      ((float*)out)[idx]  = v;
                    else if (mode == 1) ((bf16_t*)out)[idx] = (bf16_t)v;
                    else                ((float*)out)[idx] += v;
                }
            }
        }
    }
}

// ---------------------------------------------------------------------------
// Flash attention: one wave per (b, h, 16-query tile). Causal, D=64.
// qkv: bf16 [B*T, 3C] (q at +0, k at +C, v at +2C within a row)
// vT : bf16 [B, H, D, T]  (value transposed for contiguous B-fragments)
// y  : bf16 [B*T, C]
// ---------------------------------------------------------------------------
__global__ void __launch_bounds__(32)
attn_kernel(const bf16_t* __restrict__ qkv, const bf16_t* __restrict__ vT,
            bf16_t* __restrict__ y) {
    const int lane = threadIdx.x;
    const int q0   = blockIdx.x * 16;
    const int h    = blockIdx.y;
    const int b    = blockIdx.z;
    const int C3   = 3 * Cc;

    const bf16_t* qbase = qkv + (long)b * Tc * C3 + h * Dc;
    const bf16_t* kbase = qbase + Cc;
    const bf16_t* vtb   = vT + ((long)(b * Hc + h)) * Dc * Tc;

    __shared__ __align__(16) bf16_t P[16 * 32];

    v8f o[4];
    float mrun[8], lrun[8];
#pragma unroll
    for (int c = 0; c < 4; ++c)
#pragma unroll
        for (int i = 0; i < 8; ++i) o[c][i] = 0.0f;
#pragma unroll
    for (int i = 0; i < 8; ++i) { mrun[i] = -__builtin_inff(); lrun[i] = 0.0f; }

    // Query fragments for K-dim chunks d=[0..31], [32..63]
    v16bf qa0 = load_a_frag(qbase + (long)q0 * C3 + 0,  C3, lane);
    v16bf qa1 = load_a_frag(qbase + (long)q0 * C3 + 32, C3, lane);

    const int hi   = lane >> 4;
    const int colb = lane & 15;
    const int jmax = (q0 + 15) >> 5;
    const float sc = 0.125f; // 1/sqrt(64)

    for (int j = 0; j <= jmax; ++j) {
        const int k0 = j * 32;

        // scores S = Q K^T for 16 queries x 32 keys
        v8f s0, s1;
#pragma unroll
        for (int i = 0; i < 8; ++i) { s0[i] = 0.0f; s1[i] = 0.0f; }
        {
            v16bf kf;
            kf = load_b_frag(kbase + (long)k0 * C3 + 0,  C3, lane);
            s0 = wmma_bf16(qa0, kf, s0);
            kf = load_b_frag(kbase + (long)k0 * C3 + 32, C3, lane);
            s0 = wmma_bf16(qa1, kf, s0);
            kf = load_b_frag(kbase + (long)(k0 + 16) * C3 + 0,  C3, lane);
            s1 = wmma_bf16(qa0, kf, s1);
            kf = load_b_frag(kbase + (long)(k0 + 16) * C3 + 32, C3, lane);
            s1 = wmma_bf16(qa1, kf, s1);
        }

        __syncthreads(); // P reused across iterations

        float rs[8];
#pragma unroll
        for (int i = 0; i < 8; ++i) {
            const int q = q0 + i + hi * 8;
            float a0 = s0[i] * sc;
            float a1 = s1[i] * sc;
            if (k0 + colb > q)      a0 = -__builtin_inff();
            if (k0 + 16 + colb > q) a1 = -__builtin_inff();

            float mx = fmaxf(a0, a1);
#pragma unroll
            for (int m = 8; m >= 1; m >>= 1) mx = fmaxf(mx, __shfl_xor(mx, m, 32));
            const float mnew = fmaxf(mrun[i], mx);
            const float r    = __expf(mrun[i] - mnew);
            const float p0   = __expf(a0 - mnew);
            const float p1   = __expf(a1 - mnew);
            float ps = p0 + p1;
#pragma unroll
            for (int m = 8; m >= 1; m >>= 1) ps += __shfl_xor(ps, m, 32);
            lrun[i] = lrun[i] * r + ps;
            mrun[i] = mnew;
            rs[i]   = r;
            P[(i + hi * 8) * 32 + colb]      = (bf16_t)p0;
            P[(i + hi * 8) * 32 + 16 + colb] = (bf16_t)p1;
        }
        __syncthreads();

        // rescale running output
#pragma unroll
        for (int c = 0; c < 4; ++c)
#pragma unroll
            for (int i = 0; i < 8; ++i) o[c][i] *= rs[i];

        // P as A-fragment (16x32)
        v16bf pf;
        {
            const bf16_t* pp = &P[(lane & 15) * 32 + hi * 8];
            v8bf lo  = *(const v8bf*)(pp);
            v8bf hi8 = *(const v8bf*)(pp + 16);
            pf = combine16(lo, hi8);
        }

        // O += P * V   (V tile as 32x64 via vT, N chunks of 16)
#pragma unroll
        for (int c = 0; c < 4; ++c) {
            v16bf vf = load_b_frag(vtb + (long)(c * 16) * Tc + k0, Tc, lane);
            o[c] = wmma_bf16(pf, vf, o[c]);
        }
    }

    // write y = O / l
#pragma unroll
    for (int c = 0; c < 4; ++c) {
#pragma unroll
        for (int i = 0; i < 8; ++i) {
            const int q = q0 + i + hi * 8;
            float val = o[c][i] / lrun[i];
            y[((long)(b * Tc + q)) * Cc + h * Dc + c * 16 + colb] = (bf16_t)val;
        }
    }
}

// ---------------------------------------------------------------------------
// Elementwise / reduction kernels
// ---------------------------------------------------------------------------

__global__ void __launch_bounds__(256)
convert_f32_bf16_kernel(const float* __restrict__ src, bf16_t* __restrict__ dst, long n) {
    long i = (long)blockIdx.x * 256 + threadIdx.x;
    if (i < n) dst[i] = (bf16_t)src[i];
}

__global__ void __launch_bounds__(256)
embed_kernel(const int* __restrict__ idx, const float* __restrict__ wte,
             const float* __restrict__ wpe, float* __restrict__ x) {
    const int bt  = blockIdx.x;
    const int t   = bt & (Tc - 1);
    const int tok = idx[bt];
    for (int c = threadIdx.x; c < Cc; c += 256)
        x[(long)bt * Cc + c] = wte[(long)tok * Cc + c] + wpe[(long)t * Cc + c];
}

// LayerNorm over C=768, one row per block; f32 in -> bf16 out
__global__ void __launch_bounds__(256)
ln_kernel(const float* __restrict__ x, const float* __restrict__ s,
          const float* __restrict__ bias, bf16_t* __restrict__ out) {
    const int row = blockIdx.x;
    const float* xr = x + (long)row * Cc;
    __shared__ float red[256];

    float loc = 0.0f;
    for (int c = threadIdx.x; c < Cc; c += 256) loc += xr[c];
    red[threadIdx.x] = loc;
    __syncthreads();
    for (int st = 128; st > 0; st >>= 1) {
        if (threadIdx.x < st) red[threadIdx.x] += red[threadIdx.x + st];
        __syncthreads();
    }
    const float mean = red[0] / (float)Cc;
    __syncthreads();

    loc = 0.0f;
    for (int c = threadIdx.x; c < Cc; c += 256) {
        float d = xr[c] - mean;
        loc += d * d;
    }
    red[threadIdx.x] = loc;
    __syncthreads();
    for (int st = 128; st > 0; st >>= 1) {
        if (threadIdx.x < st) red[threadIdx.x] += red[threadIdx.x + st];
        __syncthreads();
    }
    const float inv = rsqrtf(red[0] / (float)Cc + 1e-5f);

    for (int c = threadIdx.x; c < Cc; c += 256)
        out[(long)row * Cc + c] = (bf16_t)((xr[c] - mean) * inv * s[c] + bias[c]);
}

// Exact GELU, f32 -> bf16
__global__ void __launch_bounds__(256)
gelu_kernel(const float* __restrict__ in, bf16_t* __restrict__ out, long n) {
    long i = (long)blockIdx.x * 256 + threadIdx.x;
    if (i < n) {
        float v = in[i];
        out[i] = (bf16_t)(0.5f * v * (1.0f + erff(v * 0.70710678118654752f)));
    }
}

// vT[b,h,d,t] = qkv[b*T+t][2C + h*D + d]
__global__ void __launch_bounds__(256)
vtrans_kernel(const bf16_t* __restrict__ qkv, bf16_t* __restrict__ vT) {
    long i = (long)blockIdx.x * 256 + threadIdx.x;
    const long total = (long)BTc * Cc;
    if (i < total) {
        int d = (int)(i & (Dc - 1));
        int h = (int)((i >> 6) % Hc);
        int t = (int)((i / Cc) & (Tc - 1));
        int b = (int)(i / ((long)Cc * Tc));
        vT[(((long)(b * Hc + h)) * Dc + d) * Tc + t] =
            qkv[((long)(b * Tc + t)) * (3 * Cc) + 2 * Cc + h * Dc + d];
    }
}

// ---------------------------------------------------------------------------
// Host-side orchestration
// ---------------------------------------------------------------------------

static inline void launch_convert(const float* src, bf16_t* dst, long n, hipStream_t s) {
    long blocks = (n + 255) / 256;
    convert_f32_bf16_kernel<<<(unsigned)blocks, 256, 0, s>>>(src, dst, n);
}

static inline void launch_gemm(const bf16_t* A, const bf16_t* W, void* out,
                               int M, int N, int K, int mode, hipStream_t s) {
    dim3 g((unsigned)(M / 32), (unsigned)((N + 63) / 64));
    gemm_bf16_kernel<<<g, 32, 0, s>>>(A, W, out, M, N, K, mode);
}

extern "C" void kernel_launch(void* const* d_in, const int* in_sizes, int n_in,
                              void* d_out, int out_size, void* d_ws, size_t ws_size,
                              hipStream_t stream) {
    (void)in_sizes; (void)n_in; (void)out_size; (void)ws_size;

    const int*   idx   = (const int*)  d_in[0];
    const float* wte   = (const float*)d_in[1];
    const float* wpe   = (const float*)d_in[2];
    const float* ln1_s = (const float*)d_in[3];
    const float* ln1_b = (const float*)d_in[4];
    const float* Wqkv  = (const float*)d_in[5];
    const float* Wo    = (const float*)d_in[6];
    const float* ln2_s = (const float*)d_in[7];
    const float* ln2_b = (const float*)d_in[8];
    const float* Wfc   = (const float*)d_in[9];
    const float* Wpr   = (const float*)d_in[10];
    const float* lnf_s = (const float*)d_in[11];
    const float* lnf_b = (const float*)d_in[12];

    // ---- carve workspace ----
    size_t off = 0;
    auto alloc = [&](size_t bytes) -> void* {
        off = (off + 255) & ~(size_t)255;
        void* p = (char*)d_ws + off;
        off += bytes;
        return p;
    };
    float*  x    = (float*) alloc((size_t)BTc * Cc * sizeof(float));       // residual stream
    bf16_t* abuf = (bf16_t*)alloc((size_t)BTc * 4 * Cc * sizeof(bf16_t));  // LN / GELU out
    bf16_t* qkvb = (bf16_t*)alloc((size_t)BTc * 3 * Cc * sizeof(bf16_t));
    bf16_t* vT   = (bf16_t*)alloc((size_t)BTc * Cc * sizeof(bf16_t));
    bf16_t* yb   = (bf16_t*)alloc((size_t)BTc * Cc * sizeof(bf16_t));
    float*  ff   = (float*) alloc((size_t)BTc * 4 * Cc * sizeof(float));   // MLP hidden f32
    bf16_t* wteb = (bf16_t*)alloc((size_t)Vc * Cc * sizeof(bf16_t));
    bf16_t* wbuf = (bf16_t*)alloc((size_t)4 * Cc * Cc * sizeof(bf16_t));   // per-layer weights

    // ---- lm_head weights (also tied embedding) to bf16 ----
    launch_convert(wte, wteb, (long)Vc * Cc, stream);

    // ---- token + position embedding ----
    embed_kernel<<<BTc, 256, 0, stream>>>(idx, wte, wpe, x);

    // ---- transformer layers ----
    for (int l = 0; l < Lc; ++l) {
        // h = LN1(x)
        ln_kernel<<<BTc, 256, 0, stream>>>(x, ln1_s + (long)l * Cc, ln1_b + (long)l * Cc, abuf);
        // qkv = h @ Wqkv^T  -> bf16
        launch_convert(Wqkv + (long)l * 3 * Cc * Cc, wbuf, (long)3 * Cc * Cc, stream);
        launch_gemm(abuf, wbuf, qkvb, BTc, 3 * Cc, Cc, /*bf16*/1, stream);
        // v transpose for attention B-fragments
        {
            long n = (long)BTc * Cc;
            vtrans_kernel<<<(unsigned)((n + 255) / 256), 256, 0, stream>>>(qkvb, vT);
        }
        // y = softmax(qk^T/sqrt(D)) v   (causal flash attention)
        {
            dim3 g(Tc / 16, Hc, Bc);
            attn_kernel<<<g, 32, 0, stream>>>(qkvb, vT, yb);
        }
        // x += y @ Wo^T
        launch_convert(Wo + (long)l * Cc * Cc, wbuf, (long)Cc * Cc, stream);
        launch_gemm(yb, wbuf, x, BTc, Cc, Cc, /*add*/2, stream);

        // h = LN2(x)
        ln_kernel<<<BTc, 256, 0, stream>>>(x, ln2_s + (long)l * Cc, ln2_b + (long)l * Cc, abuf);
        // ff = h @ Wfc^T (f32), gelu -> bf16
        launch_convert(Wfc + (long)l * 4 * Cc * Cc, wbuf, (long)4 * Cc * Cc, stream);
        launch_gemm(abuf, wbuf, ff, BTc, 4 * Cc, Cc, /*f32*/0, stream);
        {
            long n = (long)BTc * 4 * Cc;
            gelu_kernel<<<(unsigned)((n + 255) / 256), 256, 0, stream>>>(ff, abuf, n);
        }
        // x += gelu @ Wpr^T
        launch_convert(Wpr + (long)l * Cc * 4 * Cc, wbuf, (long)Cc * 4 * Cc, stream);
        launch_gemm(abuf, wbuf, x, BTc, Cc, 4 * Cc, /*add*/2, stream);
    }

    // ---- final LN + tied lm_head logits (f32 out) ----
    ln_kernel<<<BTc, 256, 0, stream>>>(x, lnf_s, lnf_b, abuf);
    launch_gemm(abuf, wteb, d_out, BTc, Vc, Cc, /*f32*/0, stream);
}
